// HGT_49349174231296
// MI455X (gfx1250) — compile-verified, matrix operand order checked by reading, not run on hardware
//
#include <hip/hip_runtime.h>
#include <hip/hip_bf16.h>
#include <math.h>

// ---------------------------------------------------------------------------
// HGT (Heterogeneous Graph Transformer) forward for MI455X / gfx1250.
// Dense projections via V_WMMA_F32_16X16X32_F16 (f16 in, f32 accumulate),
// with weights pre-packed into WMMA B-fragment order (one b128-pair load per
// K-step).  Edge softmax + scatter in fp32 with global float atomics.
// ---------------------------------------------------------------------------

typedef __attribute__((ext_vector_type(16))) _Float16 v16h;
typedef __attribute__((ext_vector_type(8)))  float    v8f;

#define HGT_C 128
#define HGT_H 4
#define HGT_D 32

__device__ __forceinline__ float hgt_gelu(float x) {
    // exact gelu: 0.5*x*(1+erf(x/sqrt(2)))
    return 0.5f * x * (1.0f + erff(x * 0.7071067811865475f));
}

// order-preserving float<->uint mapping for atomic float-max via atomicMax(u32)
__device__ __forceinline__ unsigned f2ord(float f) {
    unsigned u = __float_as_uint(f);
    return (u & 0x80000000u) ? ~u : (u | 0x80000000u);
}
__device__ __forceinline__ float ord2f(unsigned u) {
    return (u & 0x80000000u) ? __uint_as_float(u & 0x7fffffffu)
                             : __uint_as_float(~u);
}

// K index inside a 32-wide tile for (vgpr v, half h, lane-half hi8):
// ISA 16-bit A 16x32 / B 32x16 fragment packing.
__device__ __forceinline__ int wmma_kb(int v, int h, int hi8) {
    return ((v & 4) ? 16 : 0) + ((v & 3) << 1) + h + hi8;
}

// ----------------------------- fill kernels --------------------------------
__global__ void fill_f32_kernel(float* __restrict__ p, float v, long n) {
    long i = (long)blockIdx.x * blockDim.x + threadIdx.x;
    long s = (long)gridDim.x * blockDim.x;
    for (; i < n; i += s) p[i] = v;
}
__global__ void fill_u32_kernel(unsigned* __restrict__ p, unsigned v, long n) {
    long i = (long)blockIdx.x * blockDim.x + threadIdx.x;
    long s = (long)gridDim.x * blockDim.x;
    for (; i < n; i += s) p[i] = v;
}

// elementwise exact gelu (hoisted out of GEMM so erf runs once per element)
__global__ void gelu_kernel(const float* __restrict__ x, float* __restrict__ y,
                            long n4) {
    long i = (long)blockIdx.x * blockDim.x + threadIdx.x;
    long s = (long)gridDim.x * blockDim.x;
    for (; i < n4; i += s) {
        float4 v = ((const float4*)x)[i];
        v.x = hgt_gelu(v.x); v.y = hgt_gelu(v.y);
        v.z = hgt_gelu(v.z); v.w = hgt_gelu(v.w);
        ((float4*)y)[i] = v;
    }
}

// ---------------------------------------------------------------------------
// Weight pre-pack: W[128,128] f32 -> f16 WMMA B-fragments.
// Fragment index: ((colTile w)*4 + kTile t)*32 + lane, 16 halves each (32 B).
// ---------------------------------------------------------------------------
__global__ __launch_bounds__(256) void pack_w_kernel(
    const float* __restrict__ W, v16h* __restrict__ Wp)
{
    const int idx = blockIdx.x * blockDim.x + threadIdx.x;   // 0..1023
    const int w = idx >> 7, t = (idx >> 5) & 3, l = idx & 31;
    const int hi8 = (l & 16) ? 8 : 0;
    const int col = (w << 4) + (l & 15);
    v16h out;
#pragma unroll
    for (int j = 0; j < 16; ++j) {
        const int k = (t << 5) + wmma_kb(j >> 1, j & 1, hi8);
        out[j] = (_Float16)W[k * HGT_C + col];
    }
    Wp[idx] = out;
}

// Relation matrix pre-pack: R[4,32,32] f32 -> f16 B-fragments.
// Fragment index: wave wv (head = wv>>1, e0 = (wv&1)*16) * 32 + lane.
__global__ __launch_bounds__(256) void pack_r_kernel(
    const float* __restrict__ R, v16h* __restrict__ Rp)
{
    const int idx = threadIdx.x;                             // 0..255
    const int wv = idx >> 5, l = idx & 31;
    const int head = wv >> 1, e0 = (wv & 1) << 4;
    const int hi8 = (l & 16) ? 8 : 0;
    const int col = e0 + (l & 15);
    const float* Rh = R + head * (HGT_D * HGT_D);
    v16h out;
#pragma unroll
    for (int j = 0; j < 16; ++j) {
        const int kb = wmma_kb(j >> 1, j & 1, hi8);
        out[j] = (_Float16)Rh[kb * HGT_D + col];
    }
    Rp[idx] = out;
}

// ---------------------------------------------------------------------------
// Fused GEMM: Y = epi( X @ W + b ), X:[n,128] f32, Wp pre-packed f16.
// One wave = one 16x16 output tile, K swept 128 in 4 WMMA steps.
// act_out: 0=id, 1=relu.
// If skip_logit != null:  g = sigmoid(*skip_logit); y = g*y + (1-g)*skip_prev.
// ---------------------------------------------------------------------------
__global__ __launch_bounds__(256) void gemm128_kernel(
    const float* __restrict__ X, const v16h* __restrict__ Wp,
    const float* __restrict__ bias, float* __restrict__ Y, int nrows,
    int act_out,
    const float* __restrict__ skip_prev, const float* __restrict__ skip_logit)
{
    const int wv   = threadIdx.x >> 5;
    const int lane = threadIdx.x & 31;
    const int hi   = lane >> 4;
    const int lr   = lane & 15;
    const int m0   = blockIdx.x << 4;
    const int n0   = wv << 4;          // 8 waves cover all 128 columns

    int row  = m0 + lr;
    int rowc = row < nrows ? row : nrows - 1;   // clamp (dup row, store guarded)
    const float* xrow = X + (size_t)rowc * HGT_C + hi * 8;

    v8f acc = {};
#pragma unroll
    for (int kt = 0; kt < 4; ++kt) {
        const float* xp = xrow + (kt << 5);
        // A fragment: two contiguous 8-float runs (K = hi8+0..7, 16+hi8+0..7)
        const float4 x0 = *(const float4*)(xp);
        const float4 x1 = *(const float4*)(xp + 4);
        const float4 x2 = *(const float4*)(xp + 16);
        const float4 x3 = *(const float4*)(xp + 20);
        v16h a;
        a[0]=(_Float16)x0.x; a[1]=(_Float16)x0.y; a[2]=(_Float16)x0.z; a[3]=(_Float16)x0.w;
        a[4]=(_Float16)x1.x; a[5]=(_Float16)x1.y; a[6]=(_Float16)x1.z; a[7]=(_Float16)x1.w;
        a[8]=(_Float16)x2.x; a[9]=(_Float16)x2.y; a[10]=(_Float16)x2.z; a[11]=(_Float16)x2.w;
        a[12]=(_Float16)x3.x; a[13]=(_Float16)x3.y; a[14]=(_Float16)x3.z; a[15]=(_Float16)x3.w;
        const v16h b = Wp[((wv << 2) + kt) * 32 + lane];
        acc = __builtin_amdgcn_wmma_f32_16x16x32_f16(
            false, a, false, b, (short)0, acc, false, false);
    }

    const int col = n0 + lr;
    const float bv = bias ? bias[col] : 0.0f;
    float g = 1.0f, og = 0.0f;
    if (skip_logit) {
        const float s = *skip_logit;
        g  = 1.0f / (1.0f + expf(-s));
        og = 1.0f - g;
    }
#pragma unroll
    for (int v = 0; v < 8; ++v) {
        const int m = m0 + v + (hi ? 8 : 0);
        if (m < nrows) {
            float y = acc[v] + bv;
            if (act_out) y = fmaxf(y, 0.0f);
            if (skip_logit)
                y = g * y + og * skip_prev[(size_t)m * HGT_C + col];
            Y[(size_t)m * HGT_C + col] = y;
        }
    }
}

// ---------------------------------------------------------------------------
// Per-head relation projection: Y[n, h*32+e] = sum_d X[n, h*32+d] * R[h,d,e]
// (block-diagonal 4x(32x32)); one WMMA (K=32) per 16x16 tile.
// ---------------------------------------------------------------------------
__global__ __launch_bounds__(256) void relproj_kernel(
    const float* __restrict__ X, const v16h* __restrict__ Rp,
    float* __restrict__ Y, int nrows)
{
    const int wv   = threadIdx.x >> 5;
    const int lane = threadIdx.x & 31;
    const int hi   = lane >> 4;
    const int lr   = lane & 15;
    const int m0   = blockIdx.x << 4;
    const int n0   = wv << 4;
    const int head = n0 >> 5;

    int row  = m0 + lr;
    int rowc = row < nrows ? row : nrows - 1;
    const float* xp = X + (size_t)rowc * HGT_C + head * HGT_D + hi * 8;

    const float4 x0 = *(const float4*)(xp);
    const float4 x1 = *(const float4*)(xp + 4);
    const float4 x2 = *(const float4*)(xp + 16);
    const float4 x3 = *(const float4*)(xp + 20);
    v16h a;
    a[0]=(_Float16)x0.x; a[1]=(_Float16)x0.y; a[2]=(_Float16)x0.z; a[3]=(_Float16)x0.w;
    a[4]=(_Float16)x1.x; a[5]=(_Float16)x1.y; a[6]=(_Float16)x1.z; a[7]=(_Float16)x1.w;
    a[8]=(_Float16)x2.x; a[9]=(_Float16)x2.y; a[10]=(_Float16)x2.z; a[11]=(_Float16)x2.w;
    a[12]=(_Float16)x3.x; a[13]=(_Float16)x3.y; a[14]=(_Float16)x3.z; a[15]=(_Float16)x3.w;
    const v16h b = Rp[(wv << 5) + lane];

    v8f acc = {};
    acc = __builtin_amdgcn_wmma_f32_16x16x32_f16(
        false, a, false, b, (short)0, acc, false, false);

    const int col = n0 + lr;
#pragma unroll
    for (int v = 0; v < 8; ++v) {
        const int m = m0 + v + (hi ? 8 : 0);
        if (m < nrows) Y[(size_t)m * HGT_C + col] = acc[v];
    }
}

// ---------------------------------------------------------------------------
// Edge pass 1: alpha[e,h] = <q[dst,h,:], krel[src,h,:]> * p_rel[h] / sqrt(D)
//              running segment max via atomicMax on order-mapped bits.
// ---------------------------------------------------------------------------
__global__ __launch_bounds__(256) void edge_alpha_kernel(
    const float* __restrict__ qd, const float* __restrict__ krel,
    const int* __restrict__ src, const int* __restrict__ dst,
    const float* __restrict__ prel, float* __restrict__ alpha,
    unsigned* __restrict__ amax, int E)
{
    const int idx = blockIdx.x * blockDim.x + threadIdx.x;
    if (idx >= E * HGT_H) return;
    const int e = idx >> 2, h = idx & 3;
    const int s = src[e], d = dst[e];
    const float4* qp = (const float4*)(qd   + (size_t)d * HGT_C + h * HGT_D);
    const float4* kp = (const float4*)(krel + (size_t)s * HGT_C + h * HGT_D);
    float acc = 0.0f;
#pragma unroll
    for (int i = 0; i < 8; ++i) {
        const float4 a = qp[i], b = kp[i];
        acc += a.x * b.x + a.y * b.y + a.z * b.z + a.w * b.w;
    }
    const float av = acc * prel[h] * 0.17677669529663687f;  // 1/sqrt(32)
    alpha[idx] = av;
    atomicMax(&amax[(size_t)d * HGT_H + h], f2ord(av));
}

// Edge pass 2: ex = exp(alpha - amax[dst]); den[dst,h] += ex
__global__ __launch_bounds__(256) void edge_exp_kernel(
    const float* __restrict__ alpha, const unsigned* __restrict__ amax,
    const int* __restrict__ dst, float* __restrict__ exb,
    float* __restrict__ den, int E)
{
    const int idx = blockIdx.x * blockDim.x + threadIdx.x;
    if (idx >= E * HGT_H) return;
    const int e = idx >> 2, h = idx & 3;
    const int d = dst[e];
    const float m  = ord2f(amax[(size_t)d * HGT_H + h]);
    const float ex = expf(alpha[idx] - m);
    exb[idx] = ex;
    atomicAdd(&den[(size_t)d * HGT_H + h], ex);
}

// Edge pass 3: agg[dst,h,:] += (ex/(den+1e-16)) * vrel[src,h,:]
__global__ __launch_bounds__(256) void edge_scatter_kernel(
    const float* __restrict__ exb, const float* __restrict__ den,
    const float* __restrict__ vrel, const int* __restrict__ src,
    const int* __restrict__ dst, float* __restrict__ agg, int E)
{
    const int idx = blockIdx.x * blockDim.x + threadIdx.x;
    if (idx >= E * HGT_H) return;
    const int e = idx >> 2, h = idx & 3;
    const int s = src[e], d = dst[e];
    const float coef = exb[idx] / (den[(size_t)d * HGT_H + h] + 1e-16f);
    const float4* vp = (const float4*)(vrel + (size_t)s * HGT_C + h * HGT_D);
    float* ap = agg + (size_t)d * HGT_C + h * HGT_D;
#pragma unroll
    for (int i = 0; i < 8; ++i) {
        const float4 v = vp[i];
        atomicAdd(ap + 4 * i + 0, coef * v.x);
        atomicAdd(ap + 4 * i + 1, coef * v.y);
        atomicAdd(ap + 4 * i + 2, coef * v.z);
        atomicAdd(ap + 4 * i + 3, coef * v.w);
    }
}

// ---------------------------------------------------------------------------
extern "C" void kernel_launch(void* const* d_in, const int* in_sizes, int n_in,
                              void* d_out, int out_size, void* d_ws, size_t ws_size,
                              hipStream_t stream)
{
    const float* x_paper  = (const float*)d_in[0];
    const float* x_author = (const float*)d_in[1];
    const float* lin_w = (const float*)d_in[2];
    const float* lin_b = (const float*)d_in[3];
    const float* k_w   = (const float*)d_in[4];
    const float* k_b   = (const float*)d_in[5];
    const float* q_w   = (const float*)d_in[6];
    const float* q_b   = (const float*)d_in[7];
    const float* v_w   = (const float*)d_in[8];
    const float* v_b   = (const float*)d_in[9];
    const float* a_w   = (const float*)d_in[10];
    const float* a_b   = (const float*)d_in[11];
    const float* skip  = (const float*)d_in[12];
    const float* a_rel = (const float*)d_in[13];
    const float* m_rel = (const float*)d_in[14];
    const float* p_rel = (const float*)d_in[15];
    const int* srcs[3] = {(const int*)d_in[16], (const int*)d_in[18], (const int*)d_in[20]};
    const int* dsts[3] = {(const int*)d_in[17], (const int*)d_in[19], (const int*)d_in[21]};
    const int  Es[3]   = {in_sizes[16], in_sizes[18], in_sizes[20]};
    const int  etSrc[3] = {0, 1, 0};   // cites p->p, writes a->p, rev p->a
    const int  etDst[3] = {0, 0, 1};

    const int NP = in_sizes[0] / HGT_C;
    const int NA = in_sizes[1] / HGT_C;
    const int Nn[2] = {NP, NA};
    const size_t np128 = (size_t)NP * HGT_C;
    const size_t na128 = (size_t)NA * HGT_C;
    const size_t tot128 = np128 + na128;
    int maxE = Es[0];
    if (Es[1] > maxE) maxE = Es[1];
    if (Es[2] > maxE) maxE = Es[2];

    // ---- workspace layout (fp32 slots; all offsets multiples of 4) ----
    float* ws = (float*)d_ws;
    size_t o = 0;
    float* bufA = ws + o; o += tot128;               // xs after wrapper
    float* bufB = ws + o; o += tot128;               // xs after layer0
    float* kdp  = ws + o; o += tot128;               // K proj; reused as gelu buf
    float* qdp  = ws + o; o += tot128;
    float* vdp  = ws + o; o += tot128;
    float* krel = ws + o; o += np128;                // relation-projected K
    float* vrel = ws + o; o += np128;
    float* aggp = ws + o; o += tot128;               // message aggregation
    float* alph = ws + o; o += (size_t)maxE * HGT_H; // per-edge alpha
    float* exb  = ws + o; o += (size_t)maxE * HGT_H; // per-edge exp
    unsigned* amax = (unsigned*)(ws + o); o += (size_t)NP * HGT_H;
    float* den  = ws + o; o += (size_t)NP * HGT_H;
    v16h* wpack = (v16h*)(ws + o); o += 8192;        // 1024 x 32B fragments
    v16h* rpack = (v16h*)(ws + o); o += 2048;        // 256 x 32B fragments

    float* kd[2]  = {kdp, kdp + np128};
    float* qd[2]  = {qdp, qdp + np128};
    float* vd[2]  = {vdp, vdp + np128};
    float* agg[2] = {aggp, aggp + np128};
    float* gbuf   = kdp;                             // gelu(agg) scratch
    float* outp   = (float*)d_out;

    const dim3 blk(256);
    auto gemm = [&](const float* X, const float* W, const float* B, float* Y,
                    int n, int ao, const float* sp, const float* sl) {
        pack_w_kernel<<<dim3(4), blk, 0, stream>>>(W, wpack);
        gemm128_kernel<<<dim3((n + 15) / 16), blk, 0, stream>>>(
            X, wpack, B, Y, n, ao, sp, sl);
    };
    auto relproj = [&](const float* X, const float* R, float* Y, int n) {
        pack_r_kernel<<<dim3(1), blk, 0, stream>>>(R, rpack);
        relproj_kernel<<<dim3((n + 15) / 16), blk, 0, stream>>>(X, rpack, Y, n);
    };
    auto fillf = [&](float* p, float v, long n) {
        long b = (n + 255) / 256; if (b > 4096) b = 4096;
        fill_f32_kernel<<<dim3((unsigned)b), blk, 0, stream>>>(p, v, n);
    };
    auto fillu = [&](unsigned* p, unsigned v, long n) {
        long b = (n + 255) / 256; if (b > 4096) b = 4096;
        fill_u32_kernel<<<dim3((unsigned)b), blk, 0, stream>>>(p, v, n);
    };

    // ---- wrapper: per-type Linear + ReLU ----
    gemm(x_paper,  lin_w,         lin_b,       bufA,         NP, 1, nullptr, nullptr);
    gemm(x_author, lin_w + 16384, lin_b + 128, bufA + np128, NA, 1, nullptr, nullptr);

    for (int l = 0; l < 2; ++l) {
        float* cur[2];
        float* nxt[2];
        if (l == 0) { cur[0] = bufA; cur[1] = bufA + np128;
                      nxt[0] = bufB; nxt[1] = bufB + np128; }
        else        { cur[0] = bufB; cur[1] = bufB + np128;
                      nxt[0] = outp; nxt[1] = outp + np128; }

        // K / Q / V projections for both node types
        for (int t = 0; t < 2; ++t) {
            const size_t wo = (size_t)(l * 2 + t) * 16384;
            const size_t bo = (size_t)(l * 2 + t) * 128;
            gemm(cur[t], k_w + wo, k_b + bo, kd[t], Nn[t], 0, nullptr, nullptr);
            gemm(cur[t], q_w + wo, q_b + bo, qd[t], Nn[t], 0, nullptr, nullptr);
            gemm(cur[t], v_w + wo, v_b + bo, vd[t], Nn[t], 0, nullptr, nullptr);
        }

        fillf(aggp, 0.0f, (long)tot128);

        for (int r = 0; r < 3; ++r) {
            const int st = etSrc[r], dt = etDst[r], E = Es[r];
            const float* Ar = a_rel + ((size_t)l * 3 + r) * (HGT_H * HGT_D * HGT_D);
            const float* Mr = m_rel + ((size_t)l * 3 + r) * (HGT_H * HGT_D * HGT_D);
            const float* Pr = p_rel + ((size_t)l * 3 + r) * HGT_H;

            relproj(kd[st], Ar, krel, Nn[st]);
            relproj(vd[st], Mr, vrel, Nn[st]);

            fillu(amax, 0u,  (long)Nn[dt] * HGT_H);   // 0 is below ord(-inf)
            fillf(den, 0.0f, (long)Nn[dt] * HGT_H);

            const dim3 eg((E * HGT_H + 255) / 256);
            edge_alpha_kernel<<<eg, blk, 0, stream>>>(
                qd[dt], krel, srcs[r], dsts[r], Pr, alph, amax, E);
            edge_exp_kernel<<<eg, blk, 0, stream>>>(
                alph, amax, dsts[r], exb, den, E);
            edge_scatter_kernel<<<eg, blk, 0, stream>>>(
                exb, den, vrel, srcs[r], dsts[r], agg[dt], E);
        }

        // output: gelu(agg) @ a_w + a_b, blended with sigmoid(skip) gate.
        // gelu hoisted to its own pass (kd buffer is dead here).
        gelu_kernel<<<dim3(2048), blk, 0, stream>>>(aggp, gbuf, (long)(tot128 / 4));
        for (int t = 0; t < 2; ++t) {
            const size_t wo = (size_t)(l * 2 + t) * 16384;
            const size_t bo = (size_t)(l * 2 + t) * 128;
            float* gt = (t == 0) ? gbuf : gbuf + np128;
            gemm(gt, a_w + wo, a_b + bo, nxt[t], Nn[t],
                 0, cur[t], skip + l * 2 + t);
        }
    }
}
